// SwinTransformerBlock_37477884625378
// MI455X (gfx1250) — matrix operand order repaired and checked by reading, not compiled, not run on hardware
//
#include <hip/hip_runtime.h>
#include <math.h>

// ---------------------------------------------------------------------------
// Swin block for MI455X (gfx1250): all GEMMs via v_wmma_f32_16x16x32_bf16
// (fp32 accumulate), elementwise math in fp32.
// GEMM: 8 waves/block, each wave owns a 32x64 output tile (8 WMMA accums),
// block tile = 128x128. K/N are template constants so the K-loop fully
// unrolls and loads pipeline across WMMA issues.
// ---------------------------------------------------------------------------

typedef __bf16 bf16;
typedef __attribute__((ext_vector_type(16))) __bf16 v16bf;
typedef __attribute__((ext_vector_type(8)))  __bf16 v8bf;
typedef __attribute__((ext_vector_type(8)))  float  v8f;

#define BDIM   32
#define HRES   56
#define WRES   56
#define CDIM   256
#define WS     7
#define NH     8
#define HD     32
#define NWIN   49           // tokens per window
#define NPAD   64           // padded tokens per window
#define NWINDOWS 2048       // B * (56/7)^2
#define WH     (NWINDOWS*NH)
#define TOK    100352       // B*H*W = NWINDOWS*NWIN
#define HID    1024

static constexpr float ATT_SCALE = 1.0f / 33554432.0f; // HD^-5 = 32^-5 (faithful to ref)

union V16U { v16bf v; v8bf h[2]; };

// A-matrix 16x32 bf16 fragment (row-major source, stride in elements).
// lane m = lane&15; kbase = (lane>>4)*8; elems: k = kbase + {0..7, 16..23}
__device__ __forceinline__ v16bf load_frag_row(const bf16* base, int stride) {
    int lane = threadIdx.x & 31;
    const bf16* p = base + (size_t)(lane & 15) * stride + ((lane >> 4) << 3);
    V16U u;
    u.h[0] = *(const v8bf*)(p);
    u.h[1] = *(const v8bf*)(p + 16);
    return u.v;
}

// B-matrix 32x16 bf16 fragment from K-contiguous storage: element (n,k) at
// base[n*stride + k]. lane n = lane&15; kbase = (lane>>4)*8; elems: k = kbase + {0..7, 16..23}
__device__ __forceinline__ v16bf load_frag_col(const bf16* base, int stride) {
    int lane = threadIdx.x & 31;
    const bf16* p = base + (size_t)(lane & 15) * stride + ((lane >> 4) << 3);
    V16U u;
    u.h[0] = *(const v8bf*)(p);
    u.h[1] = *(const v8bf*)(p + 16);
    return u.v;
}

__device__ __forceinline__ v8f wmma_bf16(v16bf a, v16bf b, v8f c) {
    return __builtin_amdgcn_wmma_f32_16x16x32_bf16(false, a, false, b, (short)0, c,
                                                   false, false);
}

// ---------------------------------------------------------------------------
// Weight convert: W (K x N) fp32 -> WT (N x K) bf16
// ---------------------------------------------------------------------------
__global__ __launch_bounds__(256) void wconv_kernel(const float* __restrict__ w,
                                                    bf16* __restrict__ wt,
                                                    int K, int N) {
    int idx = blockIdx.x * 256 + threadIdx.x;
    if (idx < K * N) {
        int k = idx / N, n = idx - k * N;
        wt[(size_t)n * K + k] = (bf16)w[idx];
    }
}

// ---------------------------------------------------------------------------
// Zero only the pad rows/cols of the per-(window,head) Q/K/V tiles.
// Q,K: [WH][64][32] rows 49..63; V^T: [WH][32][64] cols 49..63.
// ---------------------------------------------------------------------------
__global__ __launch_bounds__(256) void pad_zero_kernel(bf16* __restrict__ q,
                                                       bf16* __restrict__ k,
                                                       bf16* __restrict__ v) {
    int wh = blockIdx.x;
    bf16* qp = q + (size_t)wh * NPAD * HD;
    bf16* kp = k + (size_t)wh * NPAD * HD;
    bf16* vp = v + (size_t)wh * HD * NPAD;
    for (int i = threadIdx.x; i < (NPAD - NWIN) * HD; i += 256) {
        int t = NWIN + i / HD;
        int d = i - (i / HD) * HD;
        qp[t * HD + d] = (bf16)0.f;
        kp[t * HD + d] = (bf16)0.f;
        vp[d * NPAD + t] = (bf16)0.f;
    }
}

// ---------------------------------------------------------------------------
// LayerNorm over C=256, one row per block (256 threads).
// MODE 0: scatter to window order (bf16).  MODE 1: row-major (bf16).
// ---------------------------------------------------------------------------
template <int MODE>
__global__ __launch_bounds__(256) void layernorm_kernel(const float* __restrict__ x,
                                                        const float* __restrict__ w,
                                                        const float* __restrict__ b,
                                                        bf16* __restrict__ out) {
    __shared__ float red1[8], red2[8];
    size_t row = blockIdx.x;
    int c = threadIdx.x;
    float v = x[row * CDIM + c];
    float s1 = v, s2 = v * v;
    #pragma unroll
    for (int o = 16; o > 0; o >>= 1) {
        s1 += __shfl_down(s1, o, 32);
        s2 += __shfl_down(s2, o, 32);
    }
    if ((c & 31) == 0) { red1[c >> 5] = s1; red2[c >> 5] = s2; }
    __syncthreads();
    float t1 = 0.f, t2 = 0.f;
    #pragma unroll
    for (int i = 0; i < 8; ++i) { t1 += red1[i]; t2 += red2[i]; }
    float mean = t1 * (1.0f / CDIM);
    float var  = t2 * (1.0f / CDIM) - mean * mean;
    float rstd = rsqrtf(var + 1e-5f);
    float y = (v - mean) * rstd * w[c] + b[c];
    if (MODE == 0) {
        int bidx = (int)(row / (HRES * WRES));
        int rem  = (int)(row - (size_t)bidx * HRES * WRES);
        int hh = rem / WRES, ww = rem - hh * WRES;
        int widx = bidx * 64 + (hh / WS) * 8 + (ww / WS);
        int t    = (hh % WS) * WS + (ww % WS);
        out[((size_t)widx * NWIN + t) * CDIM + c] = (bf16)y;
    } else {
        out[row * CDIM + c] = (bf16)y;
    }
}

// ---------------------------------------------------------------------------
// bf16 WMMA GEMM: out = A(MxK) @ WT^T + bias, with mode epilogues.
// 256 threads = 8 waves; wave (wr = wave>>1, wc = wave&1) computes a 32x64
// tile; block computes 128 rows x 128 cols. grid = (N/128, M/128).
//  MODE 0: scatter q/k into padded [WH][64][32] (bf16)
//  MODE 1: scatter v^T into padded [WH][32][64] (bf16)
//  MODE 2: out-proj: window-reverse, +bias, +residual(x), write fp32
//  MODE 3: +bias, exact GELU, bf16 row-major
//  MODE 4: +bias, +residual(row-major fp32), write fp32 row-major
// ---------------------------------------------------------------------------
template <int MODE, int K, int N>
__global__ __launch_bounds__(256) void gemm_bf16_kernel(const bf16* __restrict__ A,
                                                        const bf16* __restrict__ BT,
                                                        const float* __restrict__ bias,
                                                        void* __restrict__ out,
                                                        const float* __restrict__ resid) {
    int wave = threadIdx.x >> 5;
    int lane = threadIdx.x & 31;
    int wr = wave >> 1, wc = wave & 1;
    int m0 = blockIdx.y * 128 + wr * 32;           // 32 rows per wave
    int cb = blockIdx.x * 128 + wc * 64;           // 64 cols per wave

    const bf16* A0 = A + (size_t)m0 * K;
    const bf16* A1 = A0 + (size_t)16 * K;
    const bf16* B0 = BT + (size_t)cb * K;

    v8f acc[2][4];
    #pragma unroll
    for (int i = 0; i < 2; ++i)
        #pragma unroll
        for (int j = 0; j < 4; ++j)
            acc[i][j] = (v8f){0.f, 0.f, 0.f, 0.f, 0.f, 0.f, 0.f, 0.f};

    #pragma unroll
    for (int kk = 0; kk < K; kk += 32) {
        v16bf a0 = load_frag_row(A0 + kk, K);
        v16bf a1 = load_frag_row(A1 + kk, K);
        v16bf bf[4];
        #pragma unroll
        for (int j = 0; j < 4; ++j)
            bf[j] = load_frag_col(B0 + (size_t)j * 16 * K + kk, K);
        #pragma unroll
        for (int j = 0; j < 4; ++j) {
            acc[0][j] = wmma_bf16(a0, bf[j], acc[0][j]);
            acc[1][j] = wmma_bf16(a1, bf[j], acc[1][j]);
        }
    }

    #pragma unroll
    for (int i = 0; i < 2; ++i) {
        int mbase = m0 + i * 16 + ((lane >> 4) << 3);
        #pragma unroll
        for (int j = 0; j < 4; ++j) {
            int n = cb + j * 16 + (lane & 15);
            float bv = bias ? bias[n] : 0.f;
            #pragma unroll
            for (int r = 0; r < 8; ++r) {
                int g = mbase + r;
                float val = acc[i][j][r] + bv;
                if (MODE == 0) {            // q/k scatter: [WH][64][32]
                    int widx = g / NWIN, t = g - widx * NWIN;
                    int head = n >> 5, d = n & 31;
                    ((bf16*)out)[(((size_t)widx * NH + head) * NPAD + t) * HD + d] =
                        (bf16)val;
                } else if (MODE == 1) {     // v^T scatter: [WH][32][64]
                    int widx = g / NWIN, t = g - widx * NWIN;
                    int head = n >> 5, d = n & 31;
                    ((bf16*)out)[(((size_t)widx * NH + head) * HD + d) * NPAD + t] =
                        (bf16)val;
                } else if (MODE == 2) {     // out-proj + window reverse + residual
                    int widx = g / NWIN, t = g - widx * NWIN;
                    int bb = widx >> 6, wi = widx & 63;
                    int hh = (wi >> 3) * WS + t / WS;
                    int ww = (wi & 7) * WS + t % WS;
                    size_t idx = (((size_t)bb * HRES + hh) * WRES + ww) * CDIM + n;
                    ((float*)out)[idx] = val + resid[idx];
                } else if (MODE == 3) {     // GELU (exact), bf16
                    float ge = 0.5f * val * (1.0f + erff(val * 0.70710678118654752f));
                    ((bf16*)out)[(size_t)g * N + n] = (bf16)ge;
                } else {                    // MODE 4: final residual, fp32
                    size_t idx = (size_t)g * N + n;
                    ((float*)out)[idx] = val + resid[idx];
                }
            }
        }
    }
}

// ---------------------------------------------------------------------------
// Attention: one block per (window, head). Q,K padded [64][32]; V^T [32][64].
// Scores in LDS (fp32), masked softmax over j<49, P (bf16 LDS) @ V via WMMA.
// ---------------------------------------------------------------------------
__global__ __launch_bounds__(256) void attn_kernel(const bf16* __restrict__ qb,
                                                   const bf16* __restrict__ kb,
                                                   const bf16* __restrict__ vtb,
                                                   bf16* __restrict__ ob) {
    __shared__ __align__(16) float sc[NPAD * 65];
    __shared__ __align__(16) bf16  pb[NPAD * NPAD];

    int wh = blockIdx.x;
    const bf16* Q  = qb  + (size_t)wh * NPAD * HD;
    const bf16* Km = kb  + (size_t)wh * NPAD * HD;
    const bf16* Vt = vtb + (size_t)wh * HD * NPAD;

    int wave = threadIdx.x >> 5;
    int lane = threadIdx.x & 31;

    // ---- scores = Q @ K^T * scale : 4x4 tiles of 16x16, 2 per wave ----
    #pragma unroll
    for (int u = 0; u < 2; ++u) {
        int tt = wave + u * 8;
        int it = tt >> 2, jt = tt & 3;
        v16bf a = load_frag_row(Q + it * 16 * HD, HD);
        v16bf b = load_frag_col(Km + jt * 16 * HD, HD);
        v8f c = {0.f, 0.f, 0.f, 0.f, 0.f, 0.f, 0.f, 0.f};
        c = wmma_bf16(a, b, c);
        int n = jt * 16 + (lane & 15);
        int mb = it * 16 + ((lane >> 4) << 3);
        #pragma unroll
        for (int r = 0; r < 8; ++r)
            sc[(mb + r) * 65 + n] = c[r] * ATT_SCALE;
    }
    __syncthreads();

    // ---- softmax over valid keys j < 49 ----
    if (threadIdx.x < NPAD) {
        float* row = sc + threadIdx.x * 65;
        float mx = -3.0e38f;
        for (int j = 0; j < NWIN; ++j) mx = fmaxf(mx, row[j]);
        float s = 0.f;
        for (int j = 0; j < NWIN; ++j) { float e = __expf(row[j] - mx); row[j] = e; s += e; }
        float inv = 1.0f / s;
        bf16* pr = pb + threadIdx.x * NPAD;
        for (int j = 0; j < NPAD; ++j)
            pr[j] = (bf16)(j < NWIN ? row[j] * inv : 0.0f);
    }
    __syncthreads();

    // ---- O = P(64x64) @ V(64x32) : 4x2 tiles, one per wave ----
    int it = wave >> 1, jt = wave & 1;
    v8f c = {0.f, 0.f, 0.f, 0.f, 0.f, 0.f, 0.f, 0.f};
    #pragma unroll
    for (int kk = 0; kk < NPAD; kk += 32) {
        v16bf a = load_frag_row(pb + it * 16 * NPAD + kk, NPAD);
        v16bf b = load_frag_col(Vt + jt * 16 * NPAD + kk, NPAD);
        c = wmma_bf16(a, b, c);
    }
    int widx = wh >> 3, head = wh & 7;
    int d = jt * 16 + (lane & 15);
    int mb = it * 16 + ((lane >> 4) << 3);
    #pragma unroll
    for (int r = 0; r < 8; ++r) {
        int t = mb + r;
        if (t < NWIN)
            ob[((size_t)widx * NWIN + t) * CDIM + head * HD + d] = (bf16)c[r];
    }
}

// ---------------------------------------------------------------------------
// Host-side orchestration
// ---------------------------------------------------------------------------
extern "C" void kernel_launch(void* const* d_in, const int* in_sizes, int n_in,
                              void* d_out, int out_size, void* d_ws, size_t ws_size,
                              hipStream_t stream) {
    const float* x   = (const float*)d_in[0];
    const float* n1w = (const float*)d_in[1];
    const float* n1b = (const float*)d_in[2];
    const float* wq  = (const float*)d_in[3];
    const float* bq  = (const float*)d_in[4];
    const float* wk  = (const float*)d_in[5];
    const float* bk  = (const float*)d_in[6];
    const float* wv  = (const float*)d_in[7];
    const float* bv  = (const float*)d_in[8];
    const float* wo  = (const float*)d_in[9];
    const float* bo  = (const float*)d_in[10];
    const float* n2w = (const float*)d_in[11];
    const float* n2b = (const float*)d_in[12];
    const float* w1  = (const float*)d_in[13];
    const float* b1  = (const float*)d_in[14];
    const float* w2  = (const float*)d_in[15];
    const float* b2  = (const float*)d_in[16];

    char* ws = (char*)d_ws;
    // Workspace layout (bytes); buffers reused where lifetimes allow.
    const size_t szA   = (size_t)TOK * CDIM * 2;      // win bf16 / attn-out / LN2-out
    const size_t szQKV = (size_t)WH * NPAD * HD * 2;  // one padded q/k/v buffer
    const size_t szXO  = (size_t)TOK * CDIM * 4;      // fp32 residual stream
    const size_t szHID = (size_t)TOK * HID * 2;       // MLP hidden bf16
    const size_t oA   = 0;
    const size_t oQ   = oA + szA;
    const size_t oK   = oQ + szQKV;
    const size_t oV   = oK + szQKV;
    const size_t oXO  = oV + szQKV;
    const size_t oHID = oXO + szXO;
    const size_t oW   = oHID + szHID;
    bf16* winA = (bf16*)(ws + oA);      // window-partitioned LN1 out, later O, later h
    bf16* qbuf = (bf16*)(ws + oQ);
    bf16* kbuf = (bf16*)(ws + oK);
    bf16* vbuf = (bf16*)(ws + oV);
    float* xo  = (float*)(ws + oXO);
    bf16* hid  = (bf16*)(ws + oHID);
    bf16* wqT  = (bf16*)(ws + oW);
    bf16* wkT  = wqT + (size_t)CDIM * CDIM;
    bf16* wvT  = wkT + (size_t)CDIM * CDIM;
    bf16* woT  = wvT + (size_t)CDIM * CDIM;
    bf16* w1T  = woT + (size_t)CDIM * CDIM;
    bf16* w2T  = w1T + (size_t)CDIM * HID;

    // Zero only pad rows/cols of q/k/v (valid entries are always overwritten).
    pad_zero_kernel<<<WH, 256, 0, stream>>>(qbuf, kbuf, vbuf);

    // Weight transpose + bf16 convert.
    wconv_kernel<<<(CDIM * CDIM + 255) / 256, 256, 0, stream>>>(wq, wqT, CDIM, CDIM);
    wconv_kernel<<<(CDIM * CDIM + 255) / 256, 256, 0, stream>>>(wk, wkT, CDIM, CDIM);
    wconv_kernel<<<(CDIM * CDIM + 255) / 256, 256, 0, stream>>>(wv, wvT, CDIM, CDIM);
    wconv_kernel<<<(CDIM * CDIM + 255) / 256, 256, 0, stream>>>(wo, woT, CDIM, CDIM);
    wconv_kernel<<<(CDIM * HID  + 255) / 256, 256, 0, stream>>>(w1, w1T, CDIM, HID);
    wconv_kernel<<<(CDIM * HID  + 255) / 256, 256, 0, stream>>>(w2, w2T, HID, CDIM);

    // LN1 + window partition -> bf16.
    layernorm_kernel<0><<<TOK, 256, 0, stream>>>(x, n1w, n1b, winA);

    // QKV projections (scattered into padded per-(window,head) tiles).
    dim3 g256(CDIM / 128, TOK / 128);   // (2, 784)
    gemm_bf16_kernel<0, CDIM, CDIM><<<g256, 256, 0, stream>>>(winA, wqT, bq, qbuf, nullptr);
    gemm_bf16_kernel<0, CDIM, CDIM><<<g256, 256, 0, stream>>>(winA, wkT, bk, kbuf, nullptr);
    gemm_bf16_kernel<1, CDIM, CDIM><<<g256, 256, 0, stream>>>(winA, wvT, bv, vbuf, nullptr);

    // Windowed multi-head attention (winA reused as attention output).
    attn_kernel<<<WH, 256, 0, stream>>>(qbuf, kbuf, vbuf, winA);

    // Output projection + window reverse + residual -> xo (fp32, token order).
    gemm_bf16_kernel<2, CDIM, CDIM><<<g256, 256, 0, stream>>>(winA, woT, bo, xo, x);

    // LN2 -> bf16 (winA reused).
    layernorm_kernel<1><<<TOK, 256, 0, stream>>>(xo, n2w, n2b, winA);

    // MLP: GEMM1 + exact GELU, then GEMM2 + bias + residual -> d_out.
    dim3 g1024(HID / 128, TOK / 128);  // (8, 784)
    gemm_bf16_kernel<3, CDIM, HID><<<g1024, 256, 0, stream>>>(winA, w1T, b1, hid, nullptr);
    gemm_bf16_kernel<4, HID, CDIM><<<g256, 256, 0, stream>>>(hid, w2T, b2, d_out, xo);
}
